// DecoderRNN_47966194761812
// MI455X (gfx1250) — matrix operand order repaired and checked by reading, not compiled, not run on hardware
//
#include <hip/hip_runtime.h>
#include <hip/hip_bf16.h>

// ---------- sizes ----------
#define IZ   512      // latent
#define HH   1024     // hidden
#define GV   50257    // vocab
#define TT   1024     // steps
#define KD   (HH+IZ)  // 1536
#define G3   (3*HH)   // 3072
#define NK   (KD/32)  // 48 k-chunks

typedef __attribute__((ext_vector_type(16))) __bf16 v16bf;
typedef __attribute__((ext_vector_type(8)))  float  v8f;

__device__ __forceinline__ unsigned pack2bf(float a, float b) {
  __hip_bfloat16 ha = __float2bfloat16(a);
  __hip_bfloat16 hb = __float2bfloat16(b);
  unsigned short ra = *reinterpret_cast<unsigned short*>(&ha);
  unsigned short rb = *reinterpret_cast<unsigned short*>(&hb);
  return ((unsigned)rb << 16) | (unsigned)ra;
}

union BFfrag { unsigned u[8]; v16bf v; };

// ---------- kernel 0: zero the grid-barrier counter ----------
__global__ void init_counter(unsigned* cnt) { if (threadIdx.x == 0) *cnt = 0u; }

// ---------- kernel 1: gx_sos / gx_unk  (two 3072x1536 matvecs, wave per row) ----
// gx[j] = dot(wih[j], x) + bih[j] + (j<2048 ? bhh[j] : 0)
__global__ void precompute_gx(const float* __restrict__ z,
                              const float* __restrict__ embed_w,
                              const float* __restrict__ wih,
                              const float* __restrict__ bih,
                              const float* __restrict__ bhh,
                              float* __restrict__ gx_sos,
                              float* __restrict__ gx_unk) {
  int gtid = blockIdx.x * blockDim.x + threadIdx.x;
  int row  = gtid >> 5;            // wave per row
  int lane = threadIdx.x & 31;
  if (row >= G3) return;
  const float* wr = wih + (size_t)row * KD;
  float as = 0.f, au = 0.f;
  for (int k = lane; k < KD; k += 32) {
    float w = wr[k];
    if (k < HH) {
      as += w * fmaxf(embed_w[1 * HH + k], 0.f);   // relu(embed[SOS])
      au += w * fmaxf(embed_w[2 * HH + k], 0.f);   // relu(embed[UNK])
    } else {
      float zv = z[k - HH];
      as += w * zv; au += w * zv;
    }
  }
  #pragma unroll
  for (int off = 16; off > 0; off >>= 1) {
    as += __shfl_down(as, off, 32);
    au += __shfl_down(au, off, 32);
  }
  if (lane == 0) {
    float b = bih[row] + (row < 2 * HH ? bhh[row] : 0.f);
    gx_sos[row] = as + b;
    gx_unk[row] = au + b;
  }
}

// ---------- kernel 2: h0 = z @ z2h_w.T + z2h_b (wave per row) ----------
__global__ void compute_h0(const float* __restrict__ z,
                           const float* __restrict__ z2h_w,
                           const float* __restrict__ z2h_b,
                           float* __restrict__ h_buf) {
  int gtid = blockIdx.x * blockDim.x + threadIdx.x;
  int row  = gtid >> 5;
  int lane = threadIdx.x & 31;
  if (row >= HH) return;
  const float* wr = z2h_w + (size_t)row * IZ;
  float acc = 0.f;
  for (int k = lane; k < IZ; k += 32) acc += wr[k] * z[k];
  #pragma unroll
  for (int off = 16; off > 0; off >>= 1) acc += __shfl_down(acc, off, 32);
  if (lane == 0) h_buf[row] = acc + z2h_b[row];
}

// ---------- kernel 3: persistent cooperative GRU scan ----------
// 64 blocks x 384 threads. Block c owns h[c*16 .. c*16+16). It pins the 48
// whh rows for its chunk (r,z,n gates) in LDS for the entire 1024-step scan.
// One grid barrier per step (monotonic atomic counter).
__global__ void gru_scan(const float* __restrict__ gx_sos,
                         const float* __restrict__ gx_unk,
                         const float* __restrict__ whh,     // [3072,1024]
                         const float* __restrict__ bhh,     // [3072]
                         float* __restrict__ h_buf,         // [1024]
                         float* __restrict__ hs,            // [T,1024]
                         unsigned* __restrict__ counter) {
  extern __shared__ float smem[];
  float* w_lds  = smem;                 // 48*1024 floats (192 KB)
  float* h_lds  = smem + 48 * HH;       // 1024 floats
  float* gh_lds = h_lds + HH;           // 48 floats

  const int c   = blockIdx.x;           // chunk id 0..63
  const int tid = threadIdx.x;          // 0..383

  // preload whh slice: local row rr = g*16+i  ->  global row g*1024 + c*16 + i
  for (int e = tid; e < 48 * HH; e += 384) {
    int rr = e >> 10, k = e & (HH - 1);
    int g = rr >> 4, i = rr & 15;
    w_lds[e] = whh[(size_t)(g * HH + c * 16 + i) * HH + k];
  }
  __syncthreads();

  const int row = tid >> 3;   // 0..47
  const int seg = tid & 7;    // 8-lane K split (within a wave32)

  for (int t = 0; t < TT; ++t) {
    // broadcast h into LDS (volatile: force re-read past WGP caches)
    for (int k = tid; k < HH; k += 384)
      h_lds[k] = ((volatile const float*)h_buf)[k];
    __syncthreads();

    // 48 dot products of length 1024
    const float* wr = &w_lds[row << 10];
    float acc = 0.f;
    #pragma unroll 8
    for (int k = seg * 128; k < seg * 128 + 128; ++k)
      acc += wr[k] * h_lds[k];
    acc += __shfl_down(acc, 4, 8);
    acc += __shfl_down(acc, 2, 8);
    acc += __shfl_down(acc, 1, 8);
    if (seg == 0) gh_lds[row] = acc;
    __syncthreads();

    // GRU update for the 16 owned h elements
    const float* gx = (t == 0) ? gx_sos : gx_unk;
    if (tid < 16) {
      int i = tid, hidx = c * 16 + i;
      float ghr = gh_lds[i], ghz = gh_lds[16 + i], ghn = gh_lds[32 + i];
      float r = 1.f / (1.f + __expf(-(gx[hidx] + ghr)));
      float u = 1.f / (1.f + __expf(-(gx[HH + hidx] + ghz)));
      float n = tanhf(gx[2 * HH + hidx] + r * (ghn + bhh[2 * HH + hidx]));
      float hnew = (1.f - u) * n + u * h_lds[hidx];
      h_buf[hidx]               = hnew;
      hs[(size_t)t * HH + hidx] = hnew;
    }
    __threadfence();
    __syncthreads();
    if (tid == 0) {
      atomicAdd(counter, 1u);
      unsigned target = (unsigned)gridDim.x * (unsigned)(t + 1);
      while (atomicAdd(counter, 0u) < target) { }
    }
    __syncthreads();
    __threadfence();   // acquire: next-step h loads must see remote stores
  }
}

// ---------- kernel 4: logits = [hs|z] @ out_w.T + out_b  via bf16 WMMA ------
// Block: 256 thr (8 waves, 2Mx4N wave grid). Block tile 64(M) x 256(N); wave
// tile 32x64 (8 accumulators, 8 wmma per k-step). B tiles (fp32) staged into
// LDS with double-buffered global_load_async_to_lds_b128 (ASYNCcnt), A tiles
// as packed bf16 pairs. blockIdx.x sweeps M so the 16 co-resident M-blocks
// share each 256-column out_w slice through L2 (one HBM pass over 309 MB).
__global__ void out_gemm(const float* __restrict__ hs,     // [T,1024]
                         const float* __restrict__ z,      // [512]
                         const float* __restrict__ out_w,  // [V,1536]
                         const float* __restrict__ out_b,  // [V]
                         float* __restrict__ out) {        // [T,V]
  extern __shared__ unsigned usmem[];
  // layout (dwords): abuf0[1024] abuf1[1024] bbuf0[8192] bbuf1[8192]  = 72 KB
  unsigned* abuf0 = usmem;
  unsigned* abuf1 = usmem + 1024;
  float*    bbuf0 = (float*)(usmem + 2048);
  float*    bbuf1 = (float*)(usmem + 2048 + 8192);

  const int tid   = threadIdx.x;
  const int mbase = blockIdx.x * 64;    // M (16 blocks)
  const int nbase = blockIdx.y * 256;   // N (197 blocks)

  // --- stage A chunk (64 rows x 32 k) as bf16 pairs, o = [hs | z] ---
  auto stageA = [&](int k0, unsigned* ab) {
    #pragma unroll
    for (int s = 0; s < 4; ++s) {
      int id = s * 256 + tid;           // 1024 pair-dwords
      int ml = id >> 4, kp = id & 15;
      int trow = mbase + ml;
      int k = k0 + kp * 2;
      float f0 = (k     < HH) ? hs[(size_t)trow * HH + k]     : z[k - HH];
      float f1 = (k + 1 < HH) ? hs[(size_t)trow * HH + k + 1] : z[k + 1 - HH];
      ab[id] = pack2bf(f0, f1);
    }
  };

  // --- async-stage B chunk (256 rows x 32 k, raw fp32) into LDS ---
  auto issueB = [&](int k0, float* bb) {
    unsigned bb_base = (unsigned)(size_t)bb;   // low 32 bits = LDS byte addr
    #pragma unroll
    for (int s = 0; s < 8; ++s) {
      int id = s * 256 + tid;           // 2048 x 16B transfers
      int nl = id >> 3, q = id & 7;     // row, 16B chunk in row
      int n = nbase + nl;
      if (n < GV) {                     // EXEC-masked: OOB rows not issued
        const float* gp = out_w + (size_t)n * KD + k0 + q * 4;
        unsigned lp = bb_base + (unsigned)(nl * 128 + q * 16);
        asm volatile("global_load_async_to_lds_b128 %0, %1, off"
                     :: "v"(lp), "v"((unsigned long long)(const void*)gp)
                     : "memory");
      }
    }
  };

  const int wave = tid >> 5, lane = tid & 31;
  const int wm = wave >> 2, wn = wave & 3;     // 2 x 4 wave grid
  const int lrow = lane & 15, lhi = lane >> 4;

  v8f acc[2][4];
  #pragma unroll
  for (int f = 0; f < 2; ++f)
    #pragma unroll
    for (int j = 0; j < 4; ++j)
      acc[f][j] = (v8f){0.f,0.f,0.f,0.f,0.f,0.f,0.f,0.f};

  stageA(0, abuf0);
  issueB(0, bbuf0);

  for (int it = 0; it < NK; ++it) {
    const int cur = it & 1;
    unsigned* acur = cur ? abuf1 : abuf0;
    float*    bcur = cur ? bbuf1 : bbuf0;
    asm volatile("s_wait_asynccnt 0" ::: "memory");  // my B(cur) landed
    __syncthreads();        // all waves' B(cur)/A(cur) ready; prev reads done
    if (it + 1 < NK) {      // prefetch next chunk into the other buffers
      issueB((it + 1) * 32, cur ? bbuf0 : bbuf1);
      stageA((it + 1) * 32, cur ? abuf0 : abuf1);
    }

    // A fragments (16x32 bf16, ISA 7.12.2 layout), one per 16-row half
    BFfrag a[2];
    #pragma unroll
    for (int f = 0; f < 2; ++f) {
      const unsigned* ap = acur + (wm * 32 + f * 16 + lrow) * 16;
      #pragma unroll
      for (int i = 0; i < 8; ++i) {
        int pidx = (i < 4) ? (lhi * 4 + i) : (8 + lhi * 4 + (i - 4));
        a[f].u[i] = ap[pidx];
      }
    }

    #pragma unroll
    for (int j = 0; j < 4; ++j) {
      const float* bp = bcur + (wn * 64 + j * 16 + lrow) * 32 + lhi * 8;
      BFfrag b;                         // K = lhi*8 + 2i (+8 for i>=4) (,+1), cvt fp32->bf16
      #pragma unroll
      for (int i = 0; i < 8; ++i)
        b.u[i] = pack2bf(bp[2 * i + (i >= 4 ? 8 : 0)], bp[2 * i + 1 + (i >= 4 ? 8 : 0)]);
      #pragma unroll
      for (int f = 0; f < 2; ++f)
        acc[f][j] = __builtin_amdgcn_wmma_f32_16x16x32_bf16(
                      false, a[f].v, false, b.v, (short)0, acc[f][j],
                      false, false);
    }
  }

  // store D (+bias); C/D layout: lane<16 -> M=i, lane>=16 -> M=8+i
  #pragma unroll
  for (int f = 0; f < 2; ++f)
    #pragma unroll
    for (int j = 0; j < 4; ++j) {
      int col = nbase + wn * 64 + j * 16 + lrow;
      if (col < GV) {
        float bias = out_b[col];
        #pragma unroll
        for (int i = 0; i < 8; ++i) {
          int rowg = mbase + wm * 32 + f * 16 + lhi * 8 + i;
          out[(size_t)rowg * GV + col] = acc[f][j][i] + bias;
        }
      }
    }
}

// ---------- host launcher ----------
extern "C" void kernel_launch(void* const* d_in, const int* in_sizes, int n_in,
                              void* d_out, int out_size, void* d_ws, size_t ws_size,
                              hipStream_t stream) {
  const float* z       = (const float*)d_in[0];
  // d_in[1] = inputs (unused: word dropout makes forward token-independent)
  const float* embed_w = (const float*)d_in[2];
  const float* z2h_w   = (const float*)d_in[3];
  const float* z2h_b   = (const float*)d_in[4];
  const float* gru_wih = (const float*)d_in[5];
  const float* gru_bih = (const float*)d_in[6];
  const float* gru_whh = (const float*)d_in[7];
  const float* gru_bhh = (const float*)d_in[8];
  const float* out_w   = (const float*)d_in[9];
  const float* out_b   = (const float*)d_in[10];
  float* out = (float*)d_out;

  // workspace layout (floats)
  float* ws     = (float*)d_ws;
  float* hs     = ws;                     // T*H = 1,048,576 floats
  float* h_buf  = hs + (size_t)TT * HH;   // 1024
  float* gx_sos = h_buf + HH;             // 3072
  float* gx_unk = gx_sos + G3;            // 3072
  unsigned* cnt = (unsigned*)(gx_unk + G3);

  init_counter<<<1, 32, 0, stream>>>(cnt);
  precompute_gx<<<G3 / 8, 256, 0, stream>>>(z, embed_w, gru_wih, gru_bih,
                                            gru_bhh, gx_sos, gx_unk);
  compute_h0<<<HH / 8, 256, 0, stream>>>(z, z2h_w, z2h_b, h_buf);

  size_t scan_lds = (size_t)(48 * HH + HH + 64) * sizeof(float); // ~197 KB
  gru_scan<<<64, 384, scan_lds, stream>>>(gx_sos, gx_unk, gru_whh, gru_bhh,
                                          h_buf, hs, cnt);

  dim3 grid(TT / 64, (GV + 255) / 256);   // 16 (M) x 197 (N)
  size_t gemm_lds = (size_t)(2 * 1024 + 2 * 8192) * sizeof(unsigned); // 72 KB
  out_gemm<<<grid, 256, gemm_lds, stream>>>(hs, z, out_w, out_b, out);
}
